// MixtureOfExperts_32684701123014
// MI455X (gfx1250) — matrix operand order repaired and checked by reading, not compiled, not run on hardware
//
#include <hip/hip_runtime.h>
#include <hip/hip_bf16.h>
#include <string.h>

// ---------------------------------------------------------------------------
// MoE (B=8192, DIN=1024, HID=2048, DOUT=1024, E=8, TOPK=2) for MI455X/gfx1250
// bf16 WMMA grouped GEMMs with top-2 routing.
//  - fragments load as ds_load_b128 pairs (contiguous k = 16h + s per lane half)
//  - A tiles staged with global_load_async_to_lds_b128 (ASYNCcnt DMA path)
//  - B tile K-pair interleave via v_perm_b32
// ---------------------------------------------------------------------------

#define MOE_B    8192
#define MOE_DIN  1024
#define MOE_HID  2048
#define MOE_DOUT 1024
#define MOE_E    8
#define MOE_GHID 1024

typedef __attribute__((ext_vector_type(16))) __bf16 bf16x16;
typedef __attribute__((ext_vector_type(8)))  float  f32x8;

static __device__ __forceinline__ unsigned short f2bf(float x) {
  union { float f; unsigned int u; } c; c.f = x;
  unsigned int r = (c.u + 0x7FFFu + ((c.u >> 16) & 1u)) >> 16;  // RNE
  return (unsigned short)r;
}
static __device__ __forceinline__ float bf2f(unsigned short s) {
  union { unsigned int u; float f; } c; c.u = ((unsigned int)s) << 16;
  return c.f;
}

// LDS byte offset of a __shared__ pointer (for async-DMA LDS operand)
static __device__ __forceinline__ unsigned lds_offset(void* p) {
  return (unsigned)(unsigned long long)(
      (__attribute__((address_space(3))) char*)p);
}

// ---------------------------------------------------------------------------
// fp32 -> bf16 bulk conversion (vectorized: float4 -> ushort4), n % 4 == 0
// ---------------------------------------------------------------------------
__global__ void cvt_f32_bf16_kernel(const float4* __restrict__ in,
                                    ushort4* __restrict__ out, size_t n4) {
  size_t i = (size_t)blockIdx.x * blockDim.x + threadIdx.x;
  if (i < n4) {
    float4 f = in[i];
    ushort4 o;
    o.x = f2bf(f.x); o.y = f2bf(f.y); o.z = f2bf(f.z); o.w = f2bf(f.w);
    out[i] = o;
  }
}

__global__ void zero_f32_kernel(float4* __restrict__ p, size_t n4) {
  size_t i = (size_t)blockIdx.x * blockDim.x + threadIdx.x;
  if (i < n4) p[i] = make_float4(0.f, 0.f, 0.f, 0.f);
}

// ---------------------------------------------------------------------------
// Tiled WMMA GEMM: C[M,N] = act( A[M,K] @ W[K,N] (+ A2[M,K2] @ W2[K2,N]) + b )
//   256 threads = 8 wave32s; block tile 128x128; wave tile 64x32 (4x2 frags)
// ---------------------------------------------------------------------------
#define TM 128
#define TN 128
#define TK 32
#define LDT 40  // padded LDS row stride (elements); 80B rows keep 16B align

union FragCast { uint4 q[2]; bf16x16 v; };
union Ud4 { uint4 q; unsigned int d[4]; };

template <bool GATHER_A, bool HAS_A2, bool GATHER_A2, bool RELU, bool SCATTER>
__global__ __launch_bounds__(256) void moe_wmma_gemm(
    const unsigned short* __restrict__ A,  int lda,  int K,
    const unsigned short* __restrict__ A2, int lda2, int K2,
    const unsigned short* __restrict__ W,   // [K][N] row-major
    const unsigned short* __restrict__ W2,  // [K2][N] row-major
    const float* __restrict__ bias,
    const float* __restrict__ bias2,
    unsigned short* __restrict__ Cbf, int ldc,
    float* __restrict__ Cf32, int ldcf,
    const int* __restrict__ gatherIdx,
    const int* __restrict__ countPtr, int Mtotal,
    const float* __restrict__ gate, int expertId,
    int N) {
  __shared__ __align__(16) unsigned short As[TM * LDT];
  __shared__ __align__(16) unsigned short Bs[TN * LDT];
  __shared__ int rowIds[TM];

  int M = Mtotal;
  if (countPtr) {
    int c = *countPtr;
    M = (c < Mtotal) ? c : Mtotal;
  }
  const int m0 = blockIdx.y * TM;
  if (m0 >= M) return;
  const int n0 = blockIdx.x * TN;

  const int tid   = threadIdx.x;
  const int lane  = tid & 31;
  const int waveM = (tid >> 5) >> 2;  // 0..1
  const int waveN = (tid >> 5) & 3;   // 0..3

  if (tid < TM) {
    int slot = m0 + tid;
    int r = (slot < M) ? slot : (M - 1);
    rowIds[tid] = gatherIdx ? gatherIdx[r] : r;
  }

  f32x8 acc[4][2] = {};

  // per-thread staging coordinates
  const int aR0  = tid >> 2;            // A: row of chunk tid (second: +64)
  const int aK8  = (tid & 3) << 3;      // A: k offset (8 ushorts)
  const int bKc  = (tid >> 4) << 1;     // B: k pair base (0,2,..,30)
  const int bN8  = (tid & 15) << 3;     // B: n offset (8 ushorts)
  const int hh   = (lane & 16) ? 16 : 0;// A frag K-half select
  const int kb   = (lane & 16) ? 16 : 0;// B frag K-half select
  const int lm   = lane & 15;

  // LDS byte offsets for async DMA destinations (per-thread, loop-invariant)
  unsigned aLds0 = lds_offset(As + (aR0)      * LDT + aK8);
  unsigned aLds1 = lds_offset(As + (aR0 + 64) * LDT + aK8);
  unsigned short* bOut = Bs + bN8 * LDT + bKc;

  const int nSeg = HAS_A2 ? 2 : 1;
  for (int seg = 0; seg < nSeg; ++seg) {
    const unsigned short* Ap  = (seg == 0) ? A : A2;
    const int             ldA = (seg == 0) ? lda : lda2;
    const unsigned short* Wp  = (seg == 0) ? W : W2;
    const bool gath           = (seg == 0) ? GATHER_A : GATHER_A2;
    const int Kseg            = (seg == 0) ? K : K2;

    // per-segment gathered/direct row ids for the two A chunks
    int rowA0 = gath ? rowIds[aR0]      : (m0 + aR0);
    int rowA1 = gath ? rowIds[aR0 + 64] : (m0 + aR0 + 64);
    const unsigned short* aPtr0 = Ap + (size_t)rowA0 * ldA + aK8;
    const unsigned short* aPtr1 = Ap + (size_t)rowA1 * ldA + aK8;
    const unsigned short* wPtr  = Wp + (size_t)bKc * N + n0 + bN8;
    const size_t wStep = (size_t)TK * N;

    for (int kk = 0; kk < Kseg; kk += TK) {
      __syncthreads();
      // ---- A tile via async global->LDS DMA (ASYNCcnt, bypasses VGPRs) ----
      {
        unsigned long long ga0 = (unsigned long long)(aPtr0 + kk);
        unsigned long long ga1 = (unsigned long long)(aPtr1 + kk);
        asm volatile("global_load_async_to_lds_b128 %0, %1, off"
                     :: "v"(aLds0), "v"(ga0) : "memory");
        asm volatile("global_load_async_to_lds_b128 %0, %1, off"
                     :: "v"(aLds1), "v"(ga1) : "memory");
      }
      // ---- B tile to N-major: interleave 2 k-rows via v_perm_b32 ----
      {
        Ud4 ra, rb;
        ra.q = *(const uint4*)(wPtr);
        rb.q = *(const uint4*)(wPtr + N);
#pragma unroll
        for (int d = 0; d < 4; ++d) {
          // n = bN8+2d   : [ra.lo16, rb.lo16]
          // n = bN8+2d+1 : [ra.hi16, rb.hi16]
          unsigned lo = __builtin_amdgcn_perm(rb.d[d], ra.d[d], 0x05040100u);
          unsigned hi = __builtin_amdgcn_perm(rb.d[d], ra.d[d], 0x07060302u);
          *(unsigned int*)(bOut + (2 * d)     * LDT) = lo;
          *(unsigned int*)(bOut + (2 * d + 1) * LDT) = hi;
        }
        wPtr += wStep;
      }
      // prefetch next B K-rows (global_prefetch_b8)
      if (kk + TK < Kseg) {
        __builtin_prefetch(wPtr, 0, 1);
      }
      // publish tile: drain our async DMA group, then block barrier
      asm volatile("s_wait_asynccnt 0x0" ::: "memory");
      __syncthreads();

      // ---- fragments: 2 x ds_load_b128 each (contiguous k = 16h + s layout) ----
      bf16x16 af[4];
#pragma unroll
      for (int mi = 0; mi < 4; ++mi) {
        const unsigned short* base = As + (waveM * 64 + mi * 16 + lm) * LDT + hh;
        FragCast p;
        p.q[0] = *(const uint4*)(base);       // K = hh .. hh+7
        p.q[1] = *(const uint4*)(base + 8);   // K = hh+8 .. hh+15
        af[mi] = p.v;
      }
      bf16x16 bfr[2];
#pragma unroll
      for (int ni = 0; ni < 2; ++ni) {
        const unsigned short* base = Bs + (waveN * 32 + ni * 16 + lm) * LDT + kb;
        FragCast p;
        p.q[0] = *(const uint4*)(base);      // K = kb .. kb+7
        p.q[1] = *(const uint4*)(base + 8);  // K = kb+8 .. kb+15
        bfr[ni] = p.v;
      }
#pragma unroll
      for (int mi = 0; mi < 4; ++mi)
#pragma unroll
        for (int ni = 0; ni < 2; ++ni)
          acc[mi][ni] = __builtin_amdgcn_wmma_f32_16x16x32_bf16(
              false, af[mi], false, bfr[ni], (short)0, acc[mi][ni], false,
              false);
    }
  }

  // ---- epilogue: bias (+bias2), optional relu, bf16 store or gated scatter
#pragma unroll
  for (int mi = 0; mi < 4; ++mi) {
#pragma unroll
    for (int v = 0; v < 8; ++v) {
      int rloc = waveM * 64 + mi * 16 + ((lane & 16) ? (v + 8) : v);
      int slot = m0 + rloc;
      if (slot >= M) continue;
#pragma unroll
      for (int ni = 0; ni < 2; ++ni) {
        int col = n0 + waveN * 32 + ni * 16 + lm;
        float val = acc[mi][ni][v];
        val += bias[col];
        if (HAS_A2) val += bias2[col];
        if (RELU) val = fmaxf(val, 0.0f);
        if (SCATTER) {
          int token = rowIds[rloc];
          float g = gate[(size_t)token * MOE_E + expertId];
          atomicAdd(&Cf32[(size_t)token * ldcf + col], val * g);
        } else {
          Cbf[(size_t)slot * ldc + col] = f2bf(val);
        }
      }
    }
  }
}

// ---------------------------------------------------------------------------
// Gating stage 2: logits[t][e] = relu_h1[t] . gw2[:,e] + gb2[e]  (E=8)
// ---------------------------------------------------------------------------
__global__ void gate2_kernel(const unsigned short* __restrict__ g1,
                             const float* __restrict__ gw2,
                             const float* __restrict__ gb2,
                             float* __restrict__ logits, int Brows) {
  int gw = (blockIdx.x * blockDim.x + threadIdx.x) >> 5;  // wave index = token
  int lane = threadIdx.x & 31;
  if (gw >= Brows) return;
  float acc[MOE_E];
#pragma unroll
  for (int e = 0; e < MOE_E; ++e) acc[e] = 0.0f;
  for (int k = lane; k < MOE_GHID; k += 32) {
    float a = bf2f(g1[(size_t)gw * MOE_GHID + k]);
    const float* w = &gw2[(size_t)k * MOE_E];
#pragma unroll
    for (int e = 0; e < MOE_E; ++e) acc[e] += a * w[e];
  }
#pragma unroll
  for (int off = 16; off > 0; off >>= 1)
#pragma unroll
    for (int e = 0; e < MOE_E; ++e) acc[e] += __shfl_down(acc[e], off, 32);
  if (lane == 0) {
#pragma unroll
    for (int e = 0; e < MOE_E; ++e)
      logits[(size_t)gw * MOE_E + e] = acc[e] + gb2[e];
  }
}

// ---------------------------------------------------------------------------
// top-2 + softmax + routing (expert token lists via atomics)
// ---------------------------------------------------------------------------
__global__ void route_kernel(const float* __restrict__ logits,
                             float* __restrict__ gate,
                             int* __restrict__ counts,
                             int* __restrict__ idx, int Brows) {
  int t = blockIdx.x * blockDim.x + threadIdx.x;
  if (t >= Brows) return;
  float v[MOE_E];
#pragma unroll
  for (int e = 0; e < MOE_E; ++e) v[e] = logits[(size_t)t * MOE_E + e];
  int i1 = 0; float m1 = v[0];
#pragma unroll
  for (int e = 1; e < MOE_E; ++e) if (v[e] > m1) { m1 = v[e]; i1 = e; }
  int i2 = -1; float m2 = -1.0f / 0.0f;
#pragma unroll
  for (int e = 0; e < MOE_E; ++e) {
    if (e == i1) continue;
    if (v[e] > m2) { m2 = v[e]; i2 = e; }
  }
  float eb = __expf(m2 - m1);
  float inv = 1.0f / (1.0f + eb);
#pragma unroll
  for (int e = 0; e < MOE_E; ++e) gate[(size_t)t * MOE_E + e] = 0.0f;
  gate[(size_t)t * MOE_E + i1] = inv;
  gate[(size_t)t * MOE_E + i2] = eb * inv;
  int s1 = atomicAdd(&counts[i1], 1);
  idx[(size_t)i1 * MOE_B + s1] = t;
  int s2 = atomicAdd(&counts[i2], 1);
  idx[(size_t)i2 * MOE_B + s2] = t;
}

// ---------------------------------------------------------------------------
// host-side launch
// ---------------------------------------------------------------------------
static inline size_t align256(size_t x) { return (x + 255) & ~(size_t)255; }

extern "C" void kernel_launch(void* const* d_in, const int* in_sizes, int n_in,
                              void* d_out, int out_size, void* d_ws, size_t ws_size,
                              hipStream_t stream) {
  (void)in_sizes; (void)n_in; (void)out_size; (void)ws_size;
  const float* x   = (const float*)d_in[0];
  const float* w1  = (const float*)d_in[1];
  const float* b1  = (const float*)d_in[2];
  const float* w2  = (const float*)d_in[3];
  const float* b2  = (const float*)d_in[4];
  const float* w3  = (const float*)d_in[5];
  const float* b3  = (const float*)d_in[6];
  const float* wp  = (const float*)d_in[7];
  const float* bp  = (const float*)d_in[8];
  const float* gw1 = (const float*)d_in[9];
  const float* gb1 = (const float*)d_in[10];
  const float* gw2 = (const float*)d_in[11];
  const float* gb2 = (const float*)d_in[12];
  float* out = (float*)d_out;

  char* ws = (char*)d_ws;
  size_t off = 0;
  auto alloc = [&](size_t bytes) -> char* {
    char* p = ws + off;
    off = align256(off + bytes);
    return p;
  };
  const size_t nX   = (size_t)MOE_B * MOE_DIN;
  const size_t nG1  = (size_t)MOE_B * MOE_GHID;
  const size_t nW1  = (size_t)MOE_E * MOE_DIN * MOE_HID;
  const size_t nW2  = (size_t)MOE_E * MOE_HID * MOE_HID;
  const size_t nW3  = (size_t)MOE_E * MOE_HID * MOE_DOUT;
  const size_t nWP  = (size_t)MOE_E * MOE_DIN * MOE_HID;
  const size_t nGW1 = (size_t)MOE_DIN * MOE_GHID;
  const size_t nH   = (size_t)MOE_B * MOE_HID;

  unsigned short* xbf   = (unsigned short*)alloc(nX * 2);
  unsigned short* g1bf  = (unsigned short*)alloc(nG1 * 2);
  unsigned short* w1bf  = (unsigned short*)alloc(nW1 * 2);
  unsigned short* w2bf  = (unsigned short*)alloc(nW2 * 2);
  unsigned short* w3bf  = (unsigned short*)alloc(nW3 * 2);
  unsigned short* wpbf  = (unsigned short*)alloc(nWP * 2);
  unsigned short* gw1bf = (unsigned short*)alloc(nGW1 * 2);
  unsigned short* Hbuf  = (unsigned short*)alloc(nH * 2);
  unsigned short* Obuf  = (unsigned short*)alloc(nH * 2);
  float* logits = (float*)alloc((size_t)MOE_B * MOE_E * 4);
  float* gate   = (float*)alloc((size_t)MOE_B * MOE_E * 4);
  int*   counts = (int*)alloc(64);
  int*   idx    = (int*)alloc((size_t)MOE_E * MOE_B * 4);

  auto cvt = [&](const float* src, unsigned short* dst, size_t n) {
    size_t n4 = n / 4;
    cvt_f32_bf16_kernel<<<(int)((n4 + 255) / 256), 256, 0, stream>>>(
        (const float4*)src, (ushort4*)dst, n4);
  };
  cvt(x,   xbf,   nX);
  cvt(w1,  w1bf,  nW1);
  cvt(w2,  w2bf,  nW2);
  cvt(w3,  w3bf,  nW3);
  cvt(wp,  wpbf,  nWP);
  cvt(gw1, gw1bf, nGW1);

  {
    size_t nOut4 = (size_t)MOE_B * MOE_DOUT / 4;
    zero_f32_kernel<<<(int)((nOut4 + 255) / 256), 256, 0, stream>>>(
        (float4*)out, nOut4);
    zero_f32_kernel<<<1, 64, 0, stream>>>((float4*)counts, 4);
  }

  // Gating GEMM1: g1 = relu(x @ gw1 + gb1)
  {
    dim3 grid(MOE_GHID / TN, MOE_B / TM);
    moe_wmma_gemm<false, false, false, true, false><<<grid, 256, 0, stream>>>(
        xbf, MOE_DIN, MOE_DIN, nullptr, 0, 0, gw1bf, nullptr, gb1, nullptr,
        g1bf, MOE_GHID, nullptr, 0, nullptr, nullptr, MOE_B, nullptr, 0,
        MOE_GHID);
  }
  gate2_kernel<<<(MOE_B * 32 + 255) / 256, 256, 0, stream>>>(g1bf, gw2, gb2,
                                                             logits, MOE_B);
  route_kernel<<<(MOE_B + 255) / 256, 256, 0, stream>>>(logits, gate, counts,
                                                        idx, MOE_B);

  for (int e = 0; e < MOE_E; ++e) {
    const unsigned short* w1e = w1bf + (size_t)e * MOE_DIN * MOE_HID;
    const unsigned short* w2e = w2bf + (size_t)e * MOE_HID * MOE_HID;
    const unsigned short* w3e = w3bf + (size_t)e * MOE_HID * MOE_DOUT;
    const unsigned short* wpe = wpbf + (size_t)e * MOE_DIN * MOE_HID;
    const float* b1e = b1 + (size_t)e * MOE_HID;
    const float* b2e = b2 + (size_t)e * MOE_HID;
    const float* b3e = b3 + (size_t)e * MOE_DOUT;
    const float* bpe = bp + (size_t)e * MOE_HID;
    const int* cnt = counts + e;
    const int* ide = idx + (size_t)e * MOE_B;

    {  // E1: H = relu(x_gathered @ w1[e] + b1[e])
      dim3 grid(MOE_HID / TN, MOE_B / TM);
      moe_wmma_gemm<true, false, false, true, false><<<grid, 256, 0, stream>>>(
          xbf, MOE_DIN, MOE_DIN, nullptr, 0, 0, w1e, nullptr, b1e, nullptr,
          Hbuf, MOE_HID, nullptr, 0, ide, cnt, MOE_B, nullptr, 0, MOE_HID);
    }
    {  // E2: O = relu(H @ w2[e] + x_gathered @ wp[e] + b2[e] + bp[e])
      dim3 grid(MOE_HID / TN, MOE_B / TM);
      moe_wmma_gemm<false, true, true, true, false><<<grid, 256, 0, stream>>>(
          Hbuf, MOE_HID, MOE_HID, xbf, MOE_DIN, MOE_DIN, w2e, wpe, b2e, bpe,
          Obuf, MOE_HID, nullptr, 0, ide, cnt, MOE_B, nullptr, 0, MOE_HID);
    }
    {  // E3: out[token] += gate[token][e] * (O @ w3[e] + b3[e])
      dim3 grid(MOE_DOUT / TN, MOE_B / TM);
      moe_wmma_gemm<false, false, false, false, true><<<grid, 256, 0, stream>>>(
          Obuf, MOE_HID, MOE_HID, nullptr, 0, 0, w3e, nullptr, b3e, nullptr,
          nullptr, 0, out, MOE_DOUT, ide, cnt, MOE_B, gate, e, MOE_DOUT);
    }
  }
}
